// BinaryLinear_66640712564848
// MI455X (gfx1250) — compile-verified, matrix operand order checked by reading, not run on hardware
//
#include <hip/hip_runtime.h>

typedef __attribute__((ext_vector_type(16))) _Float16 v16h;
typedef __attribute__((ext_vector_type(8)))  _Float16 h8;
typedef __attribute__((ext_vector_type(4)))  _Float16 h4;
typedef __attribute__((ext_vector_type(8)))  float    v8f;

// ---------------------------------------------------------------------------
// Problem constants (from reference): x[4,2048,4096], W[16384,4096], bias[16384]
// ---------------------------------------------------------------------------
#define K_DIM 4096
#define N_DIM 16384
#define RED_BLOCKS 1024

// ---- CDNA5 async global->LDS copy (ASYNCcnt-tracked DMA, no VGPR staging) --
__device__ __forceinline__ void async_ld_b128(unsigned int lds_addr,
                                              const void* gaddr) {
    // VDST = LDS byte address (workgroup-relative, low 32 bits of flat addr),
    // VADDR = 64-bit global address, SADDR = off  (GV mode)
    asm volatile("global_load_async_to_lds_b128 %0, %1, off"
                 :
                 : "v"(lds_addr), "v"(gaddr)
                 : "memory");
}

#if __has_builtin(__builtin_amdgcn_s_wait_asynccnt)
#define WAIT_ASYNCCNT(n) __builtin_amdgcn_s_wait_asynccnt(n)
#else
#define WAIT_ASYNCCNT(n) asm volatile("s_wait_asynccnt %0" ::"i"(n) : "memory")
#endif

__device__ __forceinline__ unsigned int lds_addr32(const void* p) {
    // LDS aperture: flat addr[31:0] == workgroup-relative LDS byte address
    return (unsigned int)(unsigned long long)p;
}

// ---------------- Stage 1: partial sums of |W| (deterministic) --------------
__global__ __launch_bounds__(256) void absum_kernel(const float4* __restrict__ w,
                                                    float* __restrict__ partials,
                                                    int n4) {
    __shared__ float sm[256];
    float s = 0.0f;
    for (int i = blockIdx.x * blockDim.x + threadIdx.x; i < n4;
         i += gridDim.x * blockDim.x) {
        float4 v = w[i];
        s += fabsf(v.x) + fabsf(v.y) + fabsf(v.z) + fabsf(v.w);
    }
    sm[threadIdx.x] = s;
    __syncthreads();
    for (int st = 128; st > 0; st >>= 1) {
        if ((int)threadIdx.x < st) sm[threadIdx.x] += sm[threadIdx.x + st];
        __syncthreads();
    }
    if (threadIdx.x == 0) partials[blockIdx.x] = sm[0];
}

// ---------------- Stage 2: finalize gamma (single block) --------------------
__global__ __launch_bounds__(256) void gamma_kernel(const float* __restrict__ partials,
                                                    float* __restrict__ gbuf,
                                                    float count) {
    __shared__ float sm[256];
    float s = 0.0f;
    for (int i = threadIdx.x; i < RED_BLOCKS; i += 256) s += partials[i];
    sm[threadIdx.x] = s;
    __syncthreads();
    for (int st = 128; st > 0; st >>= 1) {
        if ((int)threadIdx.x < st) sm[threadIdx.x] += sm[threadIdx.x + st];
        __syncthreads();
    }
    if (threadIdx.x == 0) {
        float g = fmaxf(sm[0] / count, 1e-5f);
        gbuf[0] = g;
        gbuf[1] = 1.0f / g;
    }
}

// ---------------- Stage 3: ternary-quantize W -> f16 {-1,0,1} ---------------
__global__ __launch_bounds__(256) void quant_kernel(const float4* __restrict__ w,
                                                    h4* __restrict__ wq,
                                                    const float* __restrict__ gbuf,
                                                    int n4) {
    int i = blockIdx.x * 256 + threadIdx.x;
    if (i >= n4) return;
    const float invg = gbuf[1];
    float4 v = w[i];
    h4 o;
    o[0] = (_Float16)fminf(fmaxf(rintf(v.x * invg), -1.0f), 1.0f);
    o[1] = (_Float16)fminf(fmaxf(rintf(v.y * invg), -1.0f), 1.0f);
    o[2] = (_Float16)fminf(fmaxf(rintf(v.z * invg), -1.0f), 1.0f);
    o[3] = (_Float16)fminf(fmaxf(rintf(v.w * invg), -1.0f), 1.0f);
    wq[i] = o;
}

// ---------------- Stage 4: convert x -> f16 ---------------------------------
__global__ __launch_bounds__(256) void cvt_kernel(const float4* __restrict__ x,
                                                  h4* __restrict__ xh, int n4) {
    int i = blockIdx.x * 256 + threadIdx.x;
    if (i >= n4) return;
    float4 v = x[i];
    h4 o;
    o[0] = (_Float16)v.x;
    o[1] = (_Float16)v.y;
    o[2] = (_Float16)v.z;
    o[3] = (_Float16)v.w;
    xh[i] = o;
}

// ---------------- Stage 5: WMMA GEMM  out = gamma*(X @ Wq^T) + bias ---------
// Block tile 128(M) x 128(N), K-tile 64, double-buffered LDS fed by CDNA5
// async global->LDS DMA. 256 threads = 8 wave32s in 2(M) x 4(N) grid; each
// wave owns a 64x32 tile = 4x2 v8f WMMA accumulators.
#define TILE_K 64
#define KTILES (K_DIM / TILE_K)
#define LDA 72   // padded LDS row (9*16B stride -> conflict-free b128 reads)

__global__ __launch_bounds__(256) void bitnet_gemm_kernel(
    const _Float16* __restrict__ Xh, const _Float16* __restrict__ Wq,
    const float* __restrict__ bias, const float* __restrict__ gbuf,
    float* __restrict__ out) {
    __shared__ _Float16 As[2][128][LDA];
    __shared__ _Float16 Bs[2][128][LDA];

    const int tid   = threadIdx.x;
    const int wave  = tid >> 5;
    const int lane  = tid & 31;
    const int half  = lane >> 4;   // ISA f16 A/B layout: lanes 16-31 hold K+8
    const int r16   = lane & 15;
    const int waveM = wave & 1;    // 2 waves cover 128 M rows (64 each)
    const int waveN = wave >> 1;   // 4 waves cover 128 N cols (32 each)

    const int nBase = blockIdx.x * 128;
    const int mBase = blockIdx.y * 128;

    // cooperative tile DMA: 8 threads x 16B per 64-wide row, 32 rows/pass
    const int lr = tid >> 3;        // 0..31
    const int lc = (tid & 7) * 8;   // 0,8,...,56

    const _Float16* gA[4];
    const _Float16* gB[4];
    unsigned int    lA[2][4], lB[2][4];
#pragma unroll
    for (int p = 0; p < 4; p++) {
        const int r = lr + p * 32;
        gA[p] = Xh + (size_t)(mBase + r) * K_DIM + lc;
        gB[p] = Wq + (size_t)(nBase + r) * K_DIM + lc;
#pragma unroll
        for (int b = 0; b < 2; b++) {
            lA[b][p] = lds_addr32(&As[b][r][lc]);
            lB[b][p] = lds_addr32(&Bs[b][r][lc]);
        }
    }

    v8f acc[4][2];
#pragma unroll
    for (int i = 0; i < 4; i++)
#pragma unroll
        for (int j = 0; j < 2; j++) acc[i][j] = {};

    // prime the pipeline: tile 0 -> buffer 0 (8 async b128 ops per thread)
#pragma unroll
    for (int p = 0; p < 4; p++) {
        async_ld_b128(lA[0][p], gA[p]);
        async_ld_b128(lB[0][p], gB[p]);
        gA[p] += TILE_K;
        gB[p] += TILE_K;
    }

    for (int kt = 0; kt < KTILES; ++kt) {
        const int cur = kt & 1;
        if (kt + 1 < KTILES) {
            const int nxt = cur ^ 1;
#pragma unroll
            for (int p = 0; p < 4; p++) {
                async_ld_b128(lA[nxt][p], gA[p]);
                async_ld_b128(lB[nxt][p], gB[p]);
                gA[p] += TILE_K;
                gB[p] += TILE_K;
            }
            // async ops complete in order: <=8 outstanding means the current
            // tile's 8 ops have landed; only the new prefetch remains in flight
            WAIT_ASYNCCNT(8);
        } else {
            WAIT_ASYNCCNT(0);
        }
        __syncthreads();

#pragma unroll
        for (int kk = 0; kk < TILE_K; kk += 32) {
            v16h a[4], b[2];
#pragma unroll
            for (int i = 0; i < 4; i++) {
                const _Float16* p0 =
                    &As[cur][waveM * 64 + i * 16 + r16][kk + half * 8];
                *(h8*)&a[i]       = *(const h8*)p0;        // K = kk+half*8 ..+7
                *((h8*)&a[i] + 1) = *(const h8*)(p0 + 16); // K = kk+16+half*8
            }
#pragma unroll
            for (int j = 0; j < 2; j++) {
                const _Float16* p0 =
                    &Bs[cur][waveN * 32 + j * 16 + r16][kk + half * 8];
                *(h8*)&b[j]       = *(const h8*)p0;
                *((h8*)&b[j] + 1) = *(const h8*)(p0 + 16);
            }
#pragma unroll
            for (int i = 0; i < 4; i++)
#pragma unroll
                for (int j = 0; j < 2; j++)
                    acc[i][j] = __builtin_amdgcn_wmma_f32_16x16x32_f16(
                        false, a[i], false, b[j], (short)0, acc[i][j], false,
                        false);
        }
        // all waves done reading buf(cur) before next iteration's prefetch
        // may overwrite it
        __syncthreads();
    }

    // Epilogue: f32 C/D layout — VGPR r: lanes 0-15 -> M=r, lanes 16-31 -> M=8+r
    const float gamma = gbuf[0];
#pragma unroll
    for (int i = 0; i < 4; i++) {
#pragma unroll
        for (int j = 0; j < 2; j++) {
            const int m0 = mBase + waveM * 64 + i * 16 + half * 8;
            const int n  = nBase + waveN * 32 + j * 16 + r16;
            const float bv = bias[n];
            v8f c = acc[i][j];
#pragma unroll
            for (int r = 0; r < 8; r++)
                out[(size_t)(m0 + r) * N_DIM + n] = c[r] * gamma + bv;
        }
    }
}

// ---------------------------------------------------------------------------
extern "C" void kernel_launch(void* const* d_in, const int* in_sizes, int n_in,
                              void* d_out, int out_size, void* d_ws, size_t ws_size,
                              hipStream_t stream) {
    const float* x    = (const float*)d_in[0];
    const float* w    = (const float*)d_in[1];
    const float* bias = (const float*)d_in[2];
    float*       out  = (float*)d_out;

    const int M  = in_sizes[0] / K_DIM;   // 8192
    const int WN = in_sizes[1];           // 16384*4096
    const int XN = in_sizes[0];           // 8192*4096

    // workspace layout
    char*  ws       = (char*)d_ws;
    float* gbuf     = (float*)ws;                       // [0]=gamma, [1]=1/gamma
    float* partials = (float*)(ws + 256);               // RED_BLOCKS floats
    _Float16* xh    = (_Float16*)(ws + 16384);          // M*K f16
    _Float16* wq    = (_Float16*)(ws + 16384 + (size_t)XN * sizeof(_Float16));

    // 1) gamma = max(mean|W|, 1e-5)  (two-stage, deterministic)
    absum_kernel<<<RED_BLOCKS, 256, 0, stream>>>((const float4*)w, partials, WN / 4);
    gamma_kernel<<<1, 256, 0, stream>>>(partials, gbuf, (float)WN);

    // 2) quantize W -> f16 {-1,0,1};  3) convert x -> f16
    quant_kernel<<<(WN / 4 + 255) / 256, 256, 0, stream>>>((const float4*)w, (h4*)wq,
                                                           gbuf, WN / 4);
    cvt_kernel<<<(XN / 4 + 255) / 256, 256, 0, stream>>>((const float4*)x, (h4*)xh,
                                                         XN / 4);

    // 4) WMMA GEMM + epilogue (async-DMA double-buffered)
    dim3 grid(N_DIM / 128, M / 128);
    bitnet_gemm_kernel<<<grid, 256, 0, stream>>>(xh, wq, bias, gbuf, out);
}
